// Color_NGP_62732292325677
// MI455X (gfx1250) — matrix operand order repaired and checked by reading, not compiled, not run on hardware
//
#include <hip/hip_runtime.h>
#include <hip/hip_bf16.h>

typedef _Float16 v16h __attribute__((ext_vector_type(16)));
typedef _Float16 v8h  __attribute__((ext_vector_type(8)));
typedef float    v8f  __attribute__((ext_vector_type(8)));

#define NLEVELS 16
#define LOG2T   19
#define TSIZE   (1u << LOG2T)
#define TMASK   (TSIZE - 1u)

// Exact-GELU via Abramowitz-Stegun 7.1.26 erf (|err| < 1.5e-7, far below f16
// rounding of the activation). Branch-free: 1 v_rcp + 1 v_exp + ~12 FMA.
__device__ __forceinline__ float gelu_exact(float x) {
    float u  = __builtin_fabsf(x) * 0.70710678118654752f;   // |x|/sqrt(2)
    float t  = __builtin_amdgcn_rcpf(__builtin_fmaf(0.3275911f, u, 1.0f));
    float p  = 1.061405429f;
    p = __builtin_fmaf(p, t, -1.453152027f);
    p = __builtin_fmaf(p, t,  1.421413741f);
    p = __builtin_fmaf(p, t, -0.284496736f);
    p = __builtin_fmaf(p, t,  0.254829592f);
    p *= t;
    float e  = __builtin_amdgcn_exp2f(u * u * -1.4426950408889634f); // exp(-u^2)
    float er = __builtin_fmaf(-p, e, 1.0f);                 // erf(|x|/sqrt2)
    er = __builtin_copysignf(er, x);
    return 0.5f * x * (1.0f + er);
}

// Load a 16x32 f16 A-fragment from LDS (row-major, row stride S halves).
// CDNA5 16-bit A 16x32 layout: lanes 0-15 (M=lane) hold K halves [0..7],[16..23];
// lanes 16-31 (M=lane-16) hold K halves [8..15],[24..31].
__device__ __forceinline__ v16h load_a_frag(const _Float16* base, int rowBase,
                                            int kBase, int S, int lane) {
    int m    = lane & 15;
    int koff = (lane < 16) ? 0 : 8;
    const _Float16* r = base + (rowBase + m) * S + kBase + koff;
    v8h lo = *(const v8h*)(r);
    v8h hi = *(const v8h*)(r + 16);
    v16h a;
#pragma unroll
    for (int i = 0; i < 8; ++i) { a[i] = lo[i]; a[8 + i] = hi[i]; }
    return a;
}

// B-fragments are pre-swizzled in LDS: frag[lane*16 + i] with half i <-> K = 8*(lane/16) + i (+8 for i >= 8).
__device__ __forceinline__ v16h load_b_frag(const _Float16* fragBase, int lane) {
    const _Float16* r = fragBase + lane * 16;
    v8h lo = *(const v8h*)(r);
    v8h hi = *(const v8h*)(r + 8);
    v16h b;
#pragma unroll
    for (int i = 0; i < 8; ++i) { b[i] = lo[i]; b[8 + i] = hi[i]; }
    return b;
}

__global__ __launch_bounds__(256) void ngp_color_kernel(
    const float* __restrict__ inputs, const float* __restrict__ latent,
    const float* __restrict__ table,
    const float* __restrict__ W1, const float* __restrict__ b1,
    const float* __restrict__ W2, const float* __restrict__ b2,
    const float* __restrict__ W3, const float* __restrict__ b3,
    float* __restrict__ out)
{
    __shared__ alignas(16) _Float16 wS[14 * 512];   // swizzled B fragments
    __shared__ alignas(16) _Float16 encS[256 * 32]; // enc features, row-major
    __shared__ alignas(16) _Float16 h1S[256 * 64];
    __shared__ alignas(16) _Float16 h2S[256 * 64];

    const int tid   = threadIdx.x;
    const int lane  = tid & 31;
    const int wave  = tid >> 5;
    const int nlane = lane & 15;

    // ---- Phase 0: swizzle weights into per-lane-contiguous B fragments ----
    // frags 0-3: L1 (nt=0..3, K=32); frags 4-11: L2 (nt, ks); frags 12-13: L3 (ks)
    for (int e = tid; e < 14 * 512; e += 256) {
        int frag = e >> 9, idx = e & 511;
        int ln = idx >> 4, i = idx & 15;
        int nt, ks, layer;
        if (frag < 4)        { layer = 1; nt = frag;            ks = 0; }
        else if (frag < 12)  { layer = 2; nt = (frag - 4) >> 1; ks = (frag - 4) & 1; }
        else                 { layer = 3; nt = 0;               ks = frag - 12; }
        int klo = ks * 32 + ((ln < 16) ? 0 : 8) + ((i < 8) ? 0 : 8);
        int n   = nt * 16 + (ln & 15);
        float v;
        if (layer == 1)      v = W1[(klo + i) * 64 + n];
        else if (layer == 2) v = W2[(klo + i) * 64 + n];
        else                 v = (n < 3) ? W3[(klo + i) * 3 + n] : 0.0f;
        wS[e] = (_Float16)v;
    }

    // ---- Phase 1: hash-grid encoding, one point per thread ----
    {
        const int p  = blockIdx.x * 256 + tid;
        const int bb = p >> 15; // batch index (32768 points per batch row)
        float xs[4];
        xs[0] = (inputs[p * 3 + 0] + 1.0f) * 0.5f;
        xs[1] = (inputs[p * 3 + 1] + 1.0f) * 0.5f;
        xs[2] = (inputs[p * 3 + 2] + 1.0f) * 0.5f;
        xs[3] = (latent[bb]        + 1.0f) * 0.5f;

        const unsigned PR[4] = {1u, 2654435761u, 805459861u, 3674653429u};
        float g = 16.0f; // 1.3819^l * 16
        for (int l = 0; l < NLEVELS; ++l) {
            float scale = g - 1.0f;
            unsigned hA[4], hB[4];
            float    wA[4], wB[4];
#pragma unroll
            for (int d = 0; d < 4; ++d) {
                float pos = xs[d] * scale + 0.5f;
                float fl  = __builtin_floorf(pos);
                int   id  = (int)fl;
                float fr  = pos - fl;
                hA[d] = (unsigned)id * PR[d];
                hB[d] = (unsigned)(id + 1) * PR[d];
                wA[d] = 1.0f - fr;
                wB[d] = fr;
            }
            // pairwise-factored corner hashes/weights (dims 0-1, dims 2-3)
            unsigned h01[4], h23[4];
            float    w01[4], w23[4];
            h01[0] = hA[0] ^ hA[1]; w01[0] = wA[0] * wA[1];
            h01[1] = hB[0] ^ hA[1]; w01[1] = wB[0] * wA[1];
            h01[2] = hA[0] ^ hB[1]; w01[2] = wA[0] * wB[1];
            h01[3] = hB[0] ^ hB[1]; w01[3] = wB[0] * wB[1];
            h23[0] = hA[2] ^ hA[3]; w23[0] = wA[2] * wA[3];
            h23[1] = hB[2] ^ hA[3]; w23[1] = wB[2] * wA[3];
            h23[2] = hA[2] ^ hB[3]; w23[2] = wA[2] * wB[3];
            h23[3] = hB[2] ^ hB[3]; w23[3] = wB[2] * wB[3];

            const float2* tab2 = (const float2*)(table + (size_t)l * (size_t)(TSIZE * 2u));
            float f0 = 0.0f, f1 = 0.0f;
#pragma unroll
            for (int c = 0; c < 16; ++c) {
                unsigned h = (h01[c & 3] ^ h23[c >> 2]) & TMASK;
                float    w = w01[c & 3] * w23[c >> 2];
                float2   v = tab2[h];               // global_load_b64 gather
                f0 = __builtin_fmaf(w, v.x, f0);
                f1 = __builtin_fmaf(w, v.y, f1);
            }
            encS[tid * 32 + 2 * l]     = (_Float16)f0;
            encS[tid * 32 + 2 * l + 1] = (_Float16)f1;
            g *= 1.3819f;
        }
    }

    __syncthreads();

    // ---- Phase 2: MLP via WMMA; each wave owns two 16-row tiles ----
    const int t0 = wave * 2, t1 = wave * 2 + 1;
    const int moff = (lane & 16) ? 8 : 0;

    // Layer 1: enc[16x32] x W1[32x64] + b1, GELU -> h1S
    {
        v16h a0 = load_a_frag(encS, t0 * 16, 0, 32, lane);  // invariant over nt
        v16h a1 = load_a_frag(encS, t1 * 16, 0, 32, lane);
#pragma unroll
        for (int nt = 0; nt < 4; ++nt) {
            v16h b    = load_b_frag(&wS[nt * 512], lane);
            float bias = b1[nt * 16 + nlane];
            v8f c0, c1;
#pragma unroll
            for (int i = 0; i < 8; ++i) { c0[i] = bias; c1[i] = bias; }
            c0 = __builtin_amdgcn_wmma_f32_16x16x32_f16(false, a0, false, b, (short)0, c0, false, false);
            c1 = __builtin_amdgcn_wmma_f32_16x16x32_f16(false, a1, false, b, (short)0, c1, false, false);
#pragma unroll
            for (int r = 0; r < 8; ++r) {
                h1S[(t0 * 16 + r + moff) * 64 + nt * 16 + nlane] = (_Float16)gelu_exact(c0[r]);
                h1S[(t1 * 16 + r + moff) * 64 + nt * 16 + nlane] = (_Float16)gelu_exact(c1[r]);
            }
        }
    }
    __syncthreads();

    // Layer 2: h1[16x64] x W2[64x64] + b2, GELU -> h2S
    {
        v16h a00 = load_a_frag(h1S, t0 * 16, 0,  64, lane); // tile0, K 0..31
        v16h a01 = load_a_frag(h1S, t0 * 16, 32, 64, lane); // tile0, K 32..63
        v16h a10 = load_a_frag(h1S, t1 * 16, 0,  64, lane);
        v16h a11 = load_a_frag(h1S, t1 * 16, 32, 64, lane);
#pragma unroll
        for (int nt = 0; nt < 4; ++nt) {
            float bias = b2[nt * 16 + nlane];
            v8f c0, c1;
#pragma unroll
            for (int i = 0; i < 8; ++i) { c0[i] = bias; c1[i] = bias; }
            v16h bk0 = load_b_frag(&wS[(4 + nt * 2 + 0) * 512], lane);
            v16h bk1 = load_b_frag(&wS[(4 + nt * 2 + 1) * 512], lane);
            c0 = __builtin_amdgcn_wmma_f32_16x16x32_f16(false, a00, false, bk0, (short)0, c0, false, false);
            c1 = __builtin_amdgcn_wmma_f32_16x16x32_f16(false, a10, false, bk0, (short)0, c1, false, false);
            c0 = __builtin_amdgcn_wmma_f32_16x16x32_f16(false, a01, false, bk1, (short)0, c0, false, false);
            c1 = __builtin_amdgcn_wmma_f32_16x16x32_f16(false, a11, false, bk1, (short)0, c1, false, false);
#pragma unroll
            for (int r = 0; r < 8; ++r) {
                h2S[(t0 * 16 + r + moff) * 64 + nt * 16 + nlane] = (_Float16)gelu_exact(c0[r]);
                h2S[(t1 * 16 + r + moff) * 64 + nt * 16 + nlane] = (_Float16)gelu_exact(c1[r]);
            }
        }
    }
    __syncthreads();

    // Layer 3: h2[16x64] x W3[64x3] + b3 -> out (N-tile padded to 16)
    {
        float bias = (nlane < 3) ? b3[nlane] : 0.0f;
        v8f c0, c1;
#pragma unroll
        for (int i = 0; i < 8; ++i) { c0[i] = bias; c1[i] = bias; }
        v16h bk0 = load_b_frag(&wS[(12 + 0) * 512], lane);
        v16h bk1 = load_b_frag(&wS[(12 + 1) * 512], lane);
        v16h a00 = load_a_frag(h2S, t0 * 16, 0,  64, lane);
        v16h a01 = load_a_frag(h2S, t0 * 16, 32, 64, lane);
        v16h a10 = load_a_frag(h2S, t1 * 16, 0,  64, lane);
        v16h a11 = load_a_frag(h2S, t1 * 16, 32, 64, lane);
        c0 = __builtin_amdgcn_wmma_f32_16x16x32_f16(false, a00, false, bk0, (short)0, c0, false, false);
        c1 = __builtin_amdgcn_wmma_f32_16x16x32_f16(false, a10, false, bk0, (short)0, c1, false, false);
        c0 = __builtin_amdgcn_wmma_f32_16x16x32_f16(false, a01, false, bk1, (short)0, c0, false, false);
        c1 = __builtin_amdgcn_wmma_f32_16x16x32_f16(false, a11, false, bk1, (short)0, c1, false, false);
        if (nlane < 3) {
            const int blockBase = blockIdx.x * 256;
#pragma unroll
            for (int r = 0; r < 8; ++r) {
                out[(blockBase + t0 * 16 + r + moff) * 3 + nlane] = c0[r];
                out[(blockBase + t1 * 16 + r + moff) * 3 + nlane] = c1[r];
            }
        }
    }
}

extern "C" void kernel_launch(void* const* d_in, const int* in_sizes, int n_in,
                              void* d_out, int out_size, void* d_ws, size_t ws_size,
                              hipStream_t stream) {
    (void)in_sizes; (void)n_in; (void)d_ws; (void)ws_size; (void)out_size;
    const float* inputs = (const float*)d_in[0]; // [8,32768,3]
    const float* latent = (const float*)d_in[1]; // [8,1]
    const float* table  = (const float*)d_in[2]; // [16, 2^19, 2]
    const float* W1     = (const float*)d_in[3]; // [32,64]
    const float* b1     = (const float*)d_in[4]; // [64]
    const float* W2     = (const float*)d_in[5]; // [64,64]
    const float* b2     = (const float*)d_in[6]; // [64]
    const float* W3     = (const float*)d_in[7]; // [64,3]
    const float* b3     = (const float*)d_in[8]; // [3]
    float* out = (float*)d_out;                  // [8,32768,3]

    const int P = 8 * 32768;
    const int blocks = P / 256; // 1024
    ngp_color_kernel<<<blocks, 256, 0, stream>>>(inputs, latent, table,
                                                 W1, b1, W2, b2, W3, b3, out);
}
